// Framework_9045201125955
// MI455X (gfx1250) — matrix operand order repaired and checked by reading, not compiled, not run on hardware
//
#include <hip/hip_runtime.h>
#include <math.h>

typedef __attribute__((ext_vector_type(16))) _Float16 v16h;
typedef __attribute__((ext_vector_type(8)))  _Float16 v8h;
typedef __attribute__((ext_vector_type(4)))  _Float16 v4h;
typedef __attribute__((ext_vector_type(8)))  float    v8f;

#define G_   256
#define EMB_ 128

__device__ inline void atomicMaxFloat(float* addr, float val) {
  if (val >= 0.f) atomicMax((int*)addr, __float_as_int(val));
  else            atomicMin((unsigned int*)addr, (unsigned int)__float_as_int(val));
}

__global__ void fill_kernel(float* p, float v, long long n) {
  long long i = (long long)blockIdx.x * blockDim.x + threadIdx.x;
  long long stride = (long long)gridDim.x * blockDim.x;
  for (; i < n; i += stride) p[i] = v;
}

// ---------------------------------------------------------------------------
// C[M x N] = A[M x K] @ W[K x N]   (fp32 in, f16 WMMA, fp32 accumulate/out)
// Block: 256 threads = 8 waves = 4 row-tiles x 2 wave-cols; each wave computes
// a 16x32 output (2 WMMAs sharing one A fragment) -> block tile 64 x 64.
//
// LDS layouts are pre-swizzled into WMMA fragment order so each lane reads its
// whole fragment with two contiguous ds_load_b128:
//   AsF[tile][lane][e]  : A fragment layout (ISA 16-bit A 16x32 table)
//   WsT[n][k] (pitch 40): lane n holds K = half16*8 .. +7 and 16+half16*8 .. +7.
// Fills use float4 global loads (global_load_b128) + packed v4h LDS stores (A), scalar f16 stores (W).
// K % 32 == 0, N % 64 == 0 (K,N in {128,256} here).
// ---------------------------------------------------------------------------
__global__ __launch_bounds__(256)
void gemm_wmma_f16(const float* __restrict__ A, const float* __restrict__ W,
                   float* __restrict__ C, int M, int K, int N,
                   const int* __restrict__ rowmap)
{
  __shared__ __align__(16) _Float16 AsF[4][32][16];   // 4 KB
  __shared__ __align__(16) _Float16 WsT[64][40];      // 5 KB (pitch 40 halves)

  const int tid   = threadIdx.x;
  const int lane  = tid & 31;
  const int wave  = tid >> 5;
  const int wRow  = wave & 3;          // 0..3  (16-row tile)
  const int wCol  = wave >> 2;         // 0..1  (32-col slab)
  const int rowBase = blockIdx.x * 64;
  const int colBase = blockIdx.y * 64;
  const int half16  = lane >> 4;       // 0 or 1
  const int mn      = lane & 15;

  v8f acc0 = {};
  v8f acc1 = {};
  for (int k0 = 0; k0 < K; k0 += 32) {
    // ---- A tile 64x32: 512 float4 slots, 2 per thread, written pre-swizzled
#pragma unroll
    for (int it = 0; it < 2; ++it) {
      int flat = (tid + it * 256) * 4;       // element in 64x32 tile
      int r  = flat >> 5;                    // 0..63
      int kk = flat & 31;                    // 0,4,...,28
      int gr = rowBase + r; if (gr >= M) gr = M - 1;
      float4 f = *(const float4*)&A[(size_t)gr * K + k0 + kk];
      int laneT = (r & 15) + (((kk >> 3) & 1) << 4);
      int e0    = (kk & 7) + ((kk >> 4) << 3);
      v4h pk = { (_Float16)f.x, (_Float16)f.y, (_Float16)f.z, (_Float16)f.w };
      *(v4h*)&AsF[r >> 4][laneT][e0] = pk;
    }
    // ---- W tile 32x64 -> WsT[n][k]: 512 float4 slots, 2 per thread
#pragma unroll
    for (int it = 0; it < 2; ++it) {
      int flat = (tid + it * 256) * 4;
      int kr = flat >> 6;                    // 0..31
      int n0 = flat & 63;                    // 0,4,...,60
      float4 f = *(const float4*)&W[(size_t)(k0 + kr) * N + colBase + n0];
      WsT[n0 + 0][kr] = (_Float16)f.x;
      WsT[n0 + 1][kr] = (_Float16)f.y;
      WsT[n0 + 2][kr] = (_Float16)f.z;
      WsT[n0 + 3][kr] = (_Float16)f.w;
    }
    if (k0 + 32 < K) {                       // prefetch next A K-slab
      int gr = rowBase + (tid >> 2); if (gr >= M) gr = M - 1;
      __builtin_prefetch(&A[(size_t)gr * K + k0 + 32 + ((tid & 3) << 3)], 0, 1);
    }
    __syncthreads();

    // ---- fragments: two contiguous 16B LDS loads each, no repacking
    v8h al = *(const v8h*)&AsF[wRow][lane][0];
    v8h ah = *(const v8h*)&AsF[wRow][lane][8];
    v16h a = __builtin_shufflevector(al, ah, 0,1,2,3,4,5,6,7,8,9,10,11,12,13,14,15);

    const _Float16* b0p = &WsT[wCol * 32 + mn][half16 * 8];
    const _Float16* b1p = &WsT[wCol * 32 + 16 + mn][half16 * 8];
    v8h b0l = *(const v8h*)(b0p);
    v8h b0h = *(const v8h*)(b0p + 16);
    v8h b1l = *(const v8h*)(b1p);
    v8h b1h = *(const v8h*)(b1p + 16);
    v16h b0 = __builtin_shufflevector(b0l, b0h, 0,1,2,3,4,5,6,7,8,9,10,11,12,13,14,15);
    v16h b1 = __builtin_shufflevector(b1l, b1h, 0,1,2,3,4,5,6,7,8,9,10,11,12,13,14,15);

    acc0 = __builtin_amdgcn_wmma_f32_16x16x32_f16(false, a, false, b0,
                                                  (short)0, acc0, false, false);
    acc1 = __builtin_amdgcn_wmma_f32_16x16x32_f16(false, a, false, b1,
                                                  (short)0, acc1, false, false);
    __syncthreads();
  }
#pragma unroll
  for (int r = 0; r < 8; ++r) {              // D layout: M = r + 8*half16
    int row = rowBase + wRow * 16 + half16 * 8 + r;
    if (row < M) {
      int orow = rowmap ? rowmap[row] : row;
      int col0 = colBase + wCol * 32 + mn;
      C[(size_t)orow * N + col0]      = acc0[r];
      C[(size_t)orow * N + col0 + 16] = acc1[r];
    }
  }
}

// ---------------------------------------------------------------------------
// Cross-attention (centre queries neighbors): one wave32 per centre.
// ---------------------------------------------------------------------------
__global__ void centre_attn_kernel(const float* __restrict__ c, const float* __restrict__ n,
                                   const float* __restrict__ Kmat, const float* __restrict__ q,
                                   const float* __restrict__ counts_f, const int* __restrict__ offs,
                                   float* __restrict__ centre_gate, float* __restrict__ c_adj)
{
  int b = blockIdx.x, lane = threadIdx.x;
  int cnt = (int)(counts_f[b] + 0.5f);
  int off = offs[b];
  float sc = -INFINITY;
  if (lane < cnt) {
    const float* Kr = Kmat + (size_t)(off + lane) * EMB_;
    const float* qr = q + (size_t)b * EMB_;
    float d = 0.f;
#pragma unroll 8
    for (int i = 0; i < EMB_; ++i) d += Kr[i] * qr[i];
    sc = d * 0.088388347648318447f;    // 1/sqrt(128)
  }
  float m = sc;
  for (int s = 16; s > 0; s >>= 1) m = fmaxf(m, __shfl_xor(m, s, 32));
  float ex = (lane < cnt) ? __expf(sc - m) : 0.f;
  float ssum = ex;
  for (int s = 16; s > 0; s >>= 1) ssum += __shfl_xor(ssum, s, 32);
  float attn = ex / (ssum + 1e-16f);
  for (int g = lane; g < G_; g += 32) {
    float accv = 0.f;
    for (int j = 0; j < cnt; ++j) {
      float aj = __shfl(attn, j, 32);
      accv += aj * n[(size_t)(off + j) * G_ + g];
    }
    float gte = tanhf(accv);
    centre_gate[(size_t)b * G_ + g] = gte;
    c_adj[(size_t)b * G_ + g] = c[(size_t)b * G_ + g] * gte;
  }
}

// neighb_gate = tanh(nQ * cK[seg]); written in place over nQ; n_adj = n*gate
__global__ void neighb_gate_kernel(const float* __restrict__ n, float* __restrict__ nQ,
                                   const float* __restrict__ cK, const int* __restrict__ seg,
                                   float* __restrict__ n_adj, int total)
{
  int i = blockIdx.x * blockDim.x + threadIdx.x;
  if (i >= total) return;
  int row = i >> 8, g = i & 255;
  float gt = tanhf(nQ[i] * cK[(size_t)seg[row] * G_ + g]);
  nQ[i]    = gt;
  n_adj[i] = n[i] * gt;
}

__global__ void attn_all_kernel(const float* __restrict__ gate, const int* __restrict__ offs,
                                float* __restrict__ out, int B)
{
  int i = blockIdx.x * blockDim.x + threadIdx.x;
  if (i >= B * G_) return;
  int b = i >> 8, g = i & 255;
  out[i] = gate[(size_t)offs[b] * G_ + g];
}

// per-centre segment sums: neighb_expr, neighb_expr_in (one wave32 per centre)
__global__ void centre_sums_kernel(const float* __restrict__ c, const float* __restrict__ n,
                                   const float* __restrict__ n_adj,
                                   const float* __restrict__ counts_f, const int* __restrict__ offs,
                                   float* __restrict__ neighb_expr, float* __restrict__ neighb_expr_in)
{
  int b = blockIdx.x, lane = threadIdx.x;
  int cnt = (int)(counts_f[b] + 0.5f);
  int off = offs[b];
  float inv = 1.f / counts_f[b];
  for (int g = lane; g < G_; g += 32) {
    float s1 = 0.f, s2 = 0.f;
    for (int j = 0; j < cnt; ++j) {
      size_t r = (size_t)(off + j) * G_ + g;
      s1 += n_adj[r]; s2 += n[r];
    }
    neighb_expr[(size_t)b * G_ + g]    = s1 * inv;
    neighb_expr_in[(size_t)b * G_ + g] = c[(size_t)b * G_ + g] + s2;
  }
}

// ---------------------------------------------------------------------------
// GAT-1, centre destinations: deterministic wave32 softmax over cnt in-edges.
// ---------------------------------------------------------------------------
__global__ void gat1_centre_kernel(const float* __restrict__ h, const float* __restrict__ a_src1,
                                   const float* __restrict__ a_dst1, const float* __restrict__ a_feat,
                                   const int* __restrict__ centre_pos, const int* __restrict__ neighb_pos,
                                   const int* __restrict__ offs, const float* __restrict__ counts_f,
                                   float* __restrict__ x, float* __restrict__ ew)
{
  int b = blockIdx.x, lane = threadIdx.x;
  int cnt = (int)(counts_f[b] + 0.5f);
  int off = offs[b];
  int cnode = centre_pos[b];
  const float* hdr = h + (size_t)cnode * EMB_;
  int snode = (lane < cnt) ? neighb_pos[off + lane] : cnode;
  const float* hsr = h + (size_t)snode * EMB_;
  float alpha[4];
  float amean = 0.f;
#pragma unroll
  for (int hd = 0; hd < 4; ++hd) {
    float acc = 0.f;
#pragma unroll
    for (int i = 0; i < 32; ++i)
      acc += hsr[hd * 32 + i] * a_src1[hd * 32 + i] + hdr[hd * 32 + i] * a_dst1[hd * 32 + i];
    float v = acc > 0.f ? acc : 0.2f * acc;
    float mx = (lane < cnt) ? v : -INFINITY;
    for (int s = 16; s > 0; s >>= 1) mx = fmaxf(mx, __shfl_xor(mx, s, 32));
    float ex = (lane < cnt) ? __expf(v - mx) : 0.f;
    float sm = ex;
    for (int s = 16; s > 0; s >>= 1) sm += __shfl_xor(sm, s, 32);
    alpha[hd] = ex / (sm + 1e-16f);
    amean += 0.25f * alpha[hd];
  }
  float acc2 = 0.f;
  for (int i = 0; i < EMB_; ++i) acc2 += hsr[i] * a_feat[i] + hdr[i] * a_feat[EMB_ + i];
  float a2 = 1.f / (1.f + __expf(-acc2));
  if (lane < cnt) {
    size_t e = (size_t)(off + lane);
    ew[e * 4 + 0] = (float)snode; ew[e * 4 + 1] = (float)cnode;
    ew[e * 4 + 2] = amean;        ew[e * 4 + 3] = a2;
  }
#pragma unroll
  for (int hd = 0; hd < 4; ++hd) {
    int i = hd * 32 + lane;
    float acc = 0.f;
    for (int j = 0; j < cnt; ++j) {
      float aj = __shfl(alpha[hd], j, 32);
      int sn   = __shfl(snode, j, 32);
      acc += aj * h[(size_t)sn * EMB_ + i];
    }
    x[(size_t)cnode * EMB_ + i] = acc;
  }
}

// GAT-1, neighbor destinations: exactly one in-edge each -> alpha == 1.
__global__ void gat1_neighb_kernel(const float* __restrict__ h, const float* __restrict__ a_feat,
                                   const int* __restrict__ seg, const int* __restrict__ centre_pos,
                                   const int* __restrict__ neighb_pos,
                                   float* __restrict__ x, float* __restrict__ ew, int T)
{
  int t = blockIdx.x * blockDim.x + threadIdx.x;
  if (t >= T) return;
  int sn = centre_pos[seg[t]];
  int dn = neighb_pos[t];
  const float* hs  = h + (size_t)sn * EMB_;
  const float* hdr = h + (size_t)dn * EMB_;
  float acc2 = 0.f;
  for (int i = 0; i < EMB_; ++i) {
    x[(size_t)dn * EMB_ + i] = hs[i];
    acc2 += hs[i] * a_feat[i] + hdr[i] * a_feat[EMB_ + i];
  }
  size_t e = (size_t)T + t;      // second-half edge index
  ew[e * 4 + 0] = (float)sn; ew[e * 4 + 1] = (float)dn;
  ew[e * 4 + 2] = 1.f;       ew[e * 4 + 3] = 1.f / (1.f + __expf(-acc2));
}

__global__ void gather_xc_kernel(const float* __restrict__ x, const int* __restrict__ centre_pos,
                                 float* __restrict__ xc, int B)
{
  int i = blockIdx.x * blockDim.x + threadIdx.x;
  if (i >= B * EMB_) return;
  int b = i >> 7, d = i & 127;
  xc[i] = x[(size_t)centre_pos[b] * EMB_ + d];
}

// ----------------- virtual-tissue GAT (random edges): 3-pass softmax -------
__global__ void vt_pass1(const float* __restrict__ h2, const float* __restrict__ a_src2,
                         const float* __restrict__ a_dst2, const int* __restrict__ evt,
                         float* __restrict__ e2b, float* __restrict__ m2, int Evt)
{
  int e = blockIdx.x * blockDim.x + threadIdx.x;
  if (e >= Evt) return;
  int sv = evt[e], dv = evt[Evt + e];
  const float* hs = h2 + (size_t)sv * EMB_;
  const float* hd = h2 + (size_t)dv * EMB_;
  float acc = 0.f;
  for (int i = 0; i < EMB_; ++i) acc += hs[i] * a_src2[i] + hd[i] * a_dst2[i];
  float v = acc > 0.f ? acc : 0.2f * acc;
  e2b[e] = v;
  atomicMaxFloat(&m2[dv], v);
}
__global__ void vt_pass2(const int* __restrict__ evt, float* __restrict__ e2b,
                         const float* __restrict__ m2, float* __restrict__ s2, int Evt)
{
  int e = blockIdx.x * blockDim.x + threadIdx.x;
  if (e >= Evt) return;
  int dv = evt[Evt + e];
  float ex = __expf(e2b[e] - m2[dv]);
  e2b[e] = ex;
  atomicAdd(&s2[dv], ex);
}
__global__ void vt_pass3(const float* __restrict__ h2, const int* __restrict__ evt,
                         const float* __restrict__ e2b, const float* __restrict__ s2,
                         float* __restrict__ avt, float* __restrict__ xn, int Evt)
{
  int e = blockIdx.x * blockDim.x + threadIdx.x;
  if (e >= Evt) return;
  int sv = evt[e], dv = evt[Evt + e];
  float alpha = e2b[e] / (s2[dv] + 1e-16f);
  avt[e] = alpha;
  const float* hs = h2 + (size_t)sv * EMB_;
  for (int i = 0; i < EMB_; ++i) atomicAdd(&xn[(size_t)dv * EMB_ + i], alpha * hs[i]);
}

// ---------------------------------------------------------------------------
extern "C" void kernel_launch(void* const* d_in, const int* in_sizes, int n_in,
                              void* d_out, int out_size, void* d_ws, size_t ws_size,
                              hipStream_t stream)
{
  const float* c        = (const float*)d_in[0];
  const float* nmat     = (const float*)d_in[1];
  const float* counts_f = (const float*)d_in[2];
  const float* Wq_c     = (const float*)d_in[3];
  const float* Wk_c     = (const float*)d_in[4];
  const float* Wq_n     = (const float*)d_in[5];
  const float* Wk_n     = (const float*)d_in[6];
  const float* W1       = (const float*)d_in[7];
  const float* a_src1   = (const float*)d_in[8];
  const float* a_dst1   = (const float*)d_in[9];
  const float* a_feat   = (const float*)d_in[10];
  const float* W2       = (const float*)d_in[11];
  const float* a_src2   = (const float*)d_in[12];
  const float* a_dst2   = (const float*)d_in[13];
  const int*   seg      = (const int*)d_in[14];
  const int*   cpos     = (const int*)d_in[15];
  const int*   npos     = (const int*)d_in[16];
  const int*   offs     = (const int*)d_in[17];
  const int*   edges    = (const int*)d_in[18];
  const int*   evt      = (const int*)d_in[19];
  (void)edges; (void)n_in; (void)out_size; (void)ws_size;

  const int B   = in_sizes[2];
  const int T   = in_sizes[14];
  const int Nn  = B + T;
  const int E   = in_sizes[18] / 2;    // == 2T
  const int Evt = in_sizes[19] / 2;

  // ---- workspace carve (floats) ----
  float* ws = (float*)d_ws;
  size_t woff = 0;
  auto alloc = [&](size_t nElem) { float* p = ws + woff; woff += nElem; return p; };
  float* q    = alloc((size_t)B * EMB_);
  float* cK   = alloc((size_t)B * G_);
  float* Kb   = alloc((size_t)T * EMB_);
  float* nQ   = alloc((size_t)T * G_);     // becomes neighb_gate in place
  float* nadj = alloc((size_t)T * G_);
  float* cadj = alloc((size_t)B * G_);
  float* h    = alloc((size_t)Nn * EMB_);
  float* x    = alloc((size_t)Nn * EMB_);
  float* xc   = alloc((size_t)B * EMB_);
  float* h2   = alloc((size_t)B * EMB_);
  float* e2b  = alloc((size_t)Evt);
  float* m2   = alloc((size_t)B);
  float* s2   = alloc((size_t)B);

  // ---- output carve (reference return order) ----
  float* out      = (float*)d_out;
  float* out_xn   = out;                              // (B, EMB)
  float* out_cg   = out_xn   + (size_t)B * EMB_;      // (B, G)
  float* out_attn = out_cg   + (size_t)B * G_;        // (B, G)
  float* out_nexp = out_attn + (size_t)B * G_;        // (B, G)
  float* out_nein = out_nexp + (size_t)B * G_;        // (B, G)
  float* out_ew   = out_nein + (size_t)B * G_;        // (E, 4)
  float* out_avt  = out_ew   + (size_t)E * 4;         // (Evt,)

  // ---- init accumulators (every call; harness does not re-zero) ----
  fill_kernel<<<256, 256, 0, stream>>>(m2, -INFINITY, (long long)B);
  fill_kernel<<<256, 256, 0, stream>>>(s2, 0.f, (long long)B);
  fill_kernel<<<1024, 256, 0, stream>>>(out_xn, 0.f, (long long)B * EMB_);

  dim3 blk(256);
  auto grid = [](int M, int N) { return dim3((unsigned)((M + 63) / 64), (unsigned)(N / 64)); };

  // ---- dense GEMMs (WMMA) ----
  gemm_wmma_f16<<<grid(B, EMB_), blk, 0, stream>>>(c,    Wq_c, q,  B, G_, EMB_, nullptr);
  gemm_wmma_f16<<<grid(B, G_),   blk, 0, stream>>>(c,    Wk_n, cK, B, G_, G_,   nullptr);
  gemm_wmma_f16<<<grid(T, EMB_), blk, 0, stream>>>(nmat, Wk_c, Kb, T, G_, EMB_, nullptr);
  gemm_wmma_f16<<<grid(T, G_),   blk, 0, stream>>>(nmat, Wq_n, nQ, T, G_, G_,   nullptr);

  // ---- gates / segment ops ----
  centre_attn_kernel<<<B, 32, 0, stream>>>(c, nmat, Kb, q, counts_f, offs, out_cg, cadj);
  neighb_gate_kernel<<<(T * G_ + 255) / 256, 256, 0, stream>>>(nmat, nQ, cK, seg, nadj, T * G_);
  attn_all_kernel<<<(B * G_ + 255) / 256, 256, 0, stream>>>(nQ, offs, out_attn, B);
  centre_sums_kernel<<<B, 32, 0, stream>>>(c, nmat, nadj, counts_f, offs, out_nexp, out_nein);

  // ---- h = node_attr @ W1, scattered directly into interleaved node layout ----
  gemm_wmma_f16<<<grid(B, EMB_), blk, 0, stream>>>(cadj, W1, h, B, G_, EMB_, cpos);
  gemm_wmma_f16<<<grid(T, EMB_), blk, 0, stream>>>(nadj, W1, h, T, G_, EMB_, npos);

  // ---- GAT-1 (deterministic: structure of ragged graph is known) ----
  gat1_centre_kernel<<<B, 32, 0, stream>>>(h, a_src1, a_dst1, a_feat, cpos, npos, offs, counts_f, x, out_ew);
  gat1_neighb_kernel<<<(T + 255) / 256, 256, 0, stream>>>(h, a_feat, seg, cpos, npos, x, out_ew, T);

  // ---- virtual-tissue GAT ----
  gather_xc_kernel<<<(B * EMB_ + 255) / 256, 256, 0, stream>>>(x, cpos, xc, B);
  gemm_wmma_f16<<<grid(B, EMB_), blk, 0, stream>>>(xc, W2, h2, B, EMB_, EMB_, nullptr);
  vt_pass1<<<(Evt + 255) / 256, 256, 0, stream>>>(h2, a_src2, a_dst2, evt, e2b, m2, Evt);
  vt_pass2<<<(Evt + 255) / 256, 256, 0, stream>>>(evt, e2b, m2, s2, Evt);
  vt_pass3<<<(Evt + 255) / 256, 256, 0, stream>>>(h2, evt, e2b, s2, out_avt, out_xn, Evt);
}
